// Block_16655883174695
// MI455X (gfx1250) — compile-verified, matrix operand order checked by reading, not run on hardware
//
#include <hip/hip_runtime.h>
#include <math.h>

// ---------------------------------------------------------------------------
// Types matching gfx1250 WMMA builtin signatures
// ---------------------------------------------------------------------------
typedef __attribute__((ext_vector_type(16))) __bf16 v16bf;
typedef __attribute__((ext_vector_type(8)))  float  v8f;

#define DEV __device__ __forceinline__

// Fixed problem dims (reference: B=1, T=2048, C=1024, H=16, D=64, E=8, F=4C)
static constexpr int T_ = 2048;
static constexpr int C_ = 1024;
static constexpr int H_ = 16;
static constexpr int D_ = 64;
static constexpr int E_ = 8;
static constexpr int F_ = 4096;

DEV int lane_id() { return (int)(threadIdx.x & 31); }

// Load a 16x32 bf16 fragment (A layout; symmetric for B from a transposed
// [N][K] matrix). Source is row-major with leading dim `ld`; `tile` points at
// element [0][0] of the 16x32 tile. Pairs (k, k+1) are contiguous -> the
// compiler merges these into two global_load_b128 per fragment.
// ISA layout: lane holds row (lane&15), kbase = (lane>>4)*8,
// vgpr j<4 -> k = kbase+2j, j>=4 -> k = kbase+16+2(j-4).
// PF=true additionally issues a speculative global_prefetch_b8 two K-steps
// ahead using the SAME address register + immediate offset (no extra VGPRs;
// dropped silently if it runs past a mapped page).
template <bool PF>
DEV v16bf load_frag_t(const __bf16* tile, int ld) {
  int l  = lane_id();
  int r  = l & 15;
  int kb = (l >> 4) << 3;
  const __bf16* p = tile + (long)r * ld;
  if (PF) __builtin_prefetch((const void*)(p + 64), 0, 3);  // +128 bytes
  v16bf f;
#pragma unroll
  for (int j = 0; j < 8; ++j) {
    int k = (j < 4) ? (kb + 2 * j) : (kb + 16 + 2 * (j - 4));
    union { unsigned int u; __bf16 b[2]; } cv;
    cv.u = *(const unsigned int*)(p + k);
    f[2 * j]     = cv.b[0];
    f[2 * j + 1] = cv.b[1];
  }
  return f;
}

DEV v16bf load_frag(const __bf16* tile, int ld) {
  return load_frag_t<false>(tile, ld);
}

DEV v8f wmma_bf(v16bf a, v16bf b, v8f c) {
  return __builtin_amdgcn_wmma_f32_16x16x32_bf16(
      /*neg_a=*/false, a, /*neg_b=*/false, b,
      /*c_mod=*/(short)0, c, /*reuse_a=*/false, /*reuse_b=*/false);
}

// ---------------------------------------------------------------------------
// GEMM core: one wave computes a 32x64 tile of C = A(MxK,bf16) * BT(NxK,bf16)^T
// ---------------------------------------------------------------------------
DEV bool tile_mn(int M, int N, int& m0, int& n0) {
  int w  = (int)((blockIdx.x * blockDim.x + threadIdx.x) >> 5);
  int MT = M >> 5, NT = N >> 6;
  if (w >= MT * NT) return false;
  m0 = (w % MT) << 5;   // consecutive waves in a block share n0 -> B reuse in L2
  n0 = (w / MT) << 6;
  return true;
}

DEV void gemm_core(const __bf16* A, const __bf16* BT, int K, int m0, int n0,
                   v8f acc[2][4]) {
  for (int k = 0; k < K; k += 32) {
    v16bf a0 = load_frag_t<true>(A + (long)m0 * K + k, K);
    v16bf a1 = load_frag_t<true>(A + (long)(m0 + 16) * K + k, K);
    v16bf b0 = load_frag_t<true>(BT + (long)(n0     ) * K + k, K);
    v16bf b1 = load_frag_t<true>(BT + (long)(n0 + 16) * K + k, K);
    v16bf b2 = load_frag_t<true>(BT + (long)(n0 + 32) * K + k, K);
    v16bf b3 = load_frag_t<true>(BT + (long)(n0 + 48) * K + k, K);
    acc[0][0] = wmma_bf(a0, b0, acc[0][0]);
    acc[0][1] = wmma_bf(a0, b1, acc[0][1]);
    acc[0][2] = wmma_bf(a0, b2, acc[0][2]);
    acc[0][3] = wmma_bf(a0, b3, acc[0][3]);
    acc[1][0] = wmma_bf(a1, b0, acc[1][0]);
    acc[1][1] = wmma_bf(a1, b1, acc[1][1]);
    acc[1][2] = wmma_bf(a1, b2, acc[1][2]);
    acc[1][3] = wmma_bf(a1, b3, acc[1][3]);
  }
}

// ---------------------------------------------------------------------------
// LayerNorm: one block (256 thr) per row of C floats -> bf16 and/or f32 out
// ---------------------------------------------------------------------------
__global__ void k_ln(const float* __restrict__ x, const float* __restrict__ g,
                     const float* __restrict__ b, __bf16* outb, float* outf,
                     int C) {
  int row = blockIdx.x;
  const float* xr = x + (long)row * C;
  int tid = threadIdx.x;
  float s = 0.f, s2 = 0.f;
  for (int c = tid; c < C; c += blockDim.x) {
    float v = xr[c];
    s += v;
    s2 += v * v;
  }
#pragma unroll
  for (int m = 16; m; m >>= 1) {
    s  += __shfl_xor(s, m);
    s2 += __shfl_xor(s2, m);
  }
  __shared__ float rs[8], rs2[8];
  int w = tid >> 5;
  if ((tid & 31) == 0) { rs[w] = s; rs2[w] = s2; }
  __syncthreads();
  float ts = 0.f, t2 = 0.f;
  int nw = blockDim.x >> 5;
  for (int i = 0; i < nw; ++i) { ts += rs[i]; t2 += rs2[i]; }
  float mean = ts / C;
  float var  = t2 / C - mean * mean;
  float rinv = rsqrtf(var + 1e-5f);
  for (int c = tid; c < C; c += blockDim.x) {
    float v = (xr[c] - mean) * rinv * g[c] + b[c];
    if (outb) outb[(long)row * C + c] = (__bf16)v;
    if (outf) outf[(long)row * C + c] = v;
  }
}

// ---------------------------------------------------------------------------
// Weight conversion kernels (f32 -> bf16, transposed to [N][K])
// ---------------------------------------------------------------------------
__global__ void k_tcvt(const float* __restrict__ src, __bf16* __restrict__ dst,
                       int R, int N) {
  int i = blockIdx.x * blockDim.x + threadIdx.x;
  if (i >= R * N) return;
  int n = i / R, r = i % R;           // dst-major for coalesced writes
  dst[i] = (__bf16)src[(long)r * N + n];
}

// Fused QKV weight: WT[n'][c], n' = mat*C + h*D + d ; Wq/Wk/Wv are (H,C,D)
__global__ void k_wqkv(const float* __restrict__ Wq, const float* __restrict__ Wk,
                       const float* __restrict__ Wv, __bf16* __restrict__ WT) {
  int i = blockIdx.x * blockDim.x + threadIdx.x;
  if (i >= 3 * C_ * C_) return;
  int mat = i / (C_ * C_);
  int rem = i % (C_ * C_);
  int n = rem / C_, c = rem % C_;
  int h = n >> 6, d = n & 63;
  const float* W = (mat == 0) ? Wq : (mat == 1) ? Wk : Wv;
  WT[i] = (__bf16)W[((long)h * C_ + c) * D_ + d];
}

// ---------------------------------------------------------------------------
// GEMM 1: xn(T x C) * WqkvT -> q[h][t][d], k[h][t][d], vT[h][d][t] (bf16)
// The 64-wide, 64-aligned N-tile never crosses a matrix or head boundary,
// so mat/head are wave-uniform -> one uniform branch, no lane divergence.
// ---------------------------------------------------------------------------
__global__ void k_gemm_qkv(const __bf16* __restrict__ A,
                           const __bf16* __restrict__ BT,
                           __bf16* __restrict__ qb, __bf16* __restrict__ kb,
                           __bf16* __restrict__ vT, int M, int N, int K) {
  int m0, n0;
  if (!tile_mn(M, N, m0, n0)) return;
  v8f acc[2][4] = {};
  gemm_core(A, BT, K, m0, n0, acc);
  int l = lane_id(), cb = l & 15, rh = (l >> 4) << 3;
  int mat = n0 >> 10;                 // uniform: 0=q, 1=k, 2=v
  int hh  = (n0 >> 6) & 15;           // uniform head
  if (mat < 2) {
    __bf16* dst = (mat == 0 ? qb : kb) + (long)hh * T_ * D_;
#pragma unroll
    for (int i = 0; i < 2; ++i)
#pragma unroll
      for (int t = 0; t < 4; ++t)
#pragma unroll
        for (int r = 0; r < 8; ++r) {
          int row = m0 + 16 * i + rh + r;
          int d   = 16 * t + cb;
          dst[(long)row * D_ + d] = (__bf16)acc[i][t][r];
        }
  } else {
    __bf16* dst = vT + (long)hh * D_ * T_;
#pragma unroll
    for (int i = 0; i < 2; ++i)
#pragma unroll
      for (int t = 0; t < 4; ++t)
#pragma unroll
        for (int r = 0; r < 8; ++r) {
          int row = m0 + 16 * i + rh + r;
          int d   = 16 * t + cb;
          dst[(long)d * T_ + row] = (__bf16)acc[i][t][r];
        }
  }
}

// ---------------------------------------------------------------------------
// Flash attention: one wave per (head, 16-query tile). Causal, scale = C^-0.5
// ---------------------------------------------------------------------------
__global__ void k_attn(const __bf16* __restrict__ qb,
                       const __bf16* __restrict__ kb,
                       const __bf16* __restrict__ vT,
                       __bf16* __restrict__ attb) {
  __shared__ float plds[4][16 * 32];
  int wslot = threadIdx.x >> 5;
  float* my = &plds[wslot][0];
  int wid = blockIdx.x * (blockDim.x >> 5) + wslot;
  const int nq = T_ >> 4;
  if (wid >= H_ * nq) return;
  int h = wid / nq, q0 = (wid % nq) << 4;
  int l = lane_id(), cb = l & 15, rh = (l >> 4) << 3;

  const __bf16* qbase = qb + ((long)h * T_ + q0) * D_;
  v16bf aq0 = load_frag(qbase, D_);        // d = 0..31
  v16bf aq1 = load_frag(qbase + 32, D_);   // d = 32..63

  v8f o[4] = {};
  float mi[8], li[8];
#pragma unroll
  for (int r = 0; r < 8; ++r) { mi[r] = -__builtin_inff(); li[r] = 0.f; }
  const float scale = 0.03125f;  // 1/sqrt(C)

  for (int s0 = 0; s0 <= q0 + 15; s0 += 32) {
    v8f sc[2];
#pragma unroll
    for (int t = 0; t < 2; ++t) {
      const __bf16* kbase = kb + ((long)h * T_ + s0 + 16 * t) * D_;
      v8f s = {};
      s = wmma_bf(aq0, load_frag(kbase, D_), s);
      s = wmma_bf(aq1, load_frag(kbase + 32, D_), s);
      sc[t] = s;
    }
    // scale + causal mask (key > query -> -inf)
#pragma unroll
    for (int t = 0; t < 2; ++t)
#pragma unroll
      for (int r = 0; r < 8; ++r) {
        int key = s0 + 16 * t + cb;
        int qr  = q0 + rh + r;
        float v = sc[t][r] * scale;
        sc[t][r] = (key <= qr) ? v : -__builtin_inff();
      }
    // online softmax: per-row max/sum across the 16-lane half
    float corr[8];
#pragma unroll
    for (int r = 0; r < 8; ++r) {
      float mx = fmaxf(sc[0][r], sc[1][r]);
#pragma unroll
      for (int m = 8; m; m >>= 1) mx = fmaxf(mx, __shfl_xor(mx, m));
      float mnew = fmaxf(mi[r], mx);
      corr[r] = expf(mi[r] - mnew);
      float p0 = expf(sc[0][r] - mnew);
      float p1 = expf(sc[1][r] - mnew);
      sc[0][r] = p0; sc[1][r] = p1;
      float rsum = p0 + p1;
#pragma unroll
      for (int m = 8; m; m >>= 1) rsum += __shfl_xor(rsum, m);
      li[r] = li[r] * corr[r] + rsum;
      mi[r] = mnew;
    }
#pragma unroll
    for (int j = 0; j < 4; ++j)
#pragma unroll
      for (int r = 0; r < 8; ++r) o[j][r] *= corr[r];
    // C-layout P -> LDS -> A-fragment layout
#pragma unroll
    for (int t = 0; t < 2; ++t)
#pragma unroll
      for (int r = 0; r < 8; ++r)
        my[(rh + r) * 32 + 16 * t + cb] = sc[t][r];
    v16bf pf;
    {
      int m = l & 15, kbase = (l >> 4) << 3;
#pragma unroll
      for (int j = 0; j < 8; ++j) {
        int k = (j < 4) ? (kbase + 2 * j) : (kbase + 16 + 2 * (j - 4));
        pf[2 * j]     = (__bf16)my[m * 32 + k];
        pf[2 * j + 1] = (__bf16)my[m * 32 + k + 1];
      }
    }
#pragma unroll
    for (int j = 0; j < 4; ++j) {
      const __bf16* vbase = vT + ((long)h * D_ + 16 * j) * T_ + s0;
      o[j] = wmma_bf(pf, load_frag(vbase, T_), o[j]);
    }
  }
  // normalize + write concat-head layout attb[t][h*64 + d]
#pragma unroll
  for (int j = 0; j < 4; ++j)
#pragma unroll
    for (int r = 0; r < 8; ++r) {
      int row = q0 + rh + r;
      int col = 16 * j + cb;
      attb[(long)row * C_ + h * D_ + col] = (__bf16)(o[j][r] / li[r]);
    }
}

// ---------------------------------------------------------------------------
// GEMM 2: x2 = x + att @ Wproj + bproj  (writes f32 x2 and seeds d_out)
// ---------------------------------------------------------------------------
__global__ void k_gemm_proj(const __bf16* __restrict__ A,
                            const __bf16* __restrict__ BT,
                            const float* __restrict__ bias,
                            const float* __restrict__ resid,
                            float* x2f, float* outf, int M, int N, int K) {
  int m0, n0;
  if (!tile_mn(M, N, m0, n0)) return;
  v8f acc[2][4] = {};
  gemm_core(A, BT, K, m0, n0, acc);
  int l = lane_id(), cb = l & 15, rh = (l >> 4) << 3;
#pragma unroll
  for (int i = 0; i < 2; ++i)
#pragma unroll
    for (int t = 0; t < 4; ++t)
#pragma unroll
      for (int r = 0; r < 8; ++r) {
        int row = m0 + 16 * i + rh + r;
        int col = n0 + 16 * t + cb;
        long idx = (long)row * N + col;
        float v = acc[i][t][r] + bias[col] + resid[idx];
        x2f[idx] = v;
        outf[idx] = v;
      }
}

// GEMM 3: h1 = relu(xn2 @ W1[e] + b1[e])  -> bf16
__global__ void k_gemm_relu(const __bf16* __restrict__ A,
                            const __bf16* __restrict__ BT,
                            const float* __restrict__ bias,
                            __bf16* out, int M, int N, int K) {
  int m0, n0;
  if (!tile_mn(M, N, m0, n0)) return;
  v8f acc[2][4] = {};
  gemm_core(A, BT, K, m0, n0, acc);
  int l = lane_id(), cb = l & 15, rh = (l >> 4) << 3;
#pragma unroll
  for (int i = 0; i < 2; ++i)
#pragma unroll
    for (int t = 0; t < 4; ++t)
#pragma unroll
      for (int r = 0; r < 8; ++r) {
        int row = m0 + 16 * i + rh + r;
        int col = n0 + 16 * t + cb;
        float v = acc[i][t][r] + bias[col];
        out[(long)row * N + col] = (__bf16)(v > 0.f ? v : 0.f);
      }
}

// GEMM 4: d_out += gate[:,e] * (h1 @ W2[e] + b2[e])
__global__ void k_gemm_moe2(const __bf16* __restrict__ A,
                            const __bf16* __restrict__ BT,
                            const float* __restrict__ bias,
                            const float* __restrict__ gate, int e,
                            float* out, int M, int N, int K) {
  int m0, n0;
  if (!tile_mn(M, N, m0, n0)) return;
  v8f acc[2][4] = {};
  gemm_core(A, BT, K, m0, n0, acc);
  int l = lane_id(), cb = l & 15, rh = (l >> 4) << 3;
#pragma unroll
  for (int i = 0; i < 2; ++i)
#pragma unroll
    for (int t = 0; t < 4; ++t)
#pragma unroll
      for (int r = 0; r < 8; ++r) {
        int row = m0 + 16 * i + rh + r;
        int col = n0 + 16 * t + cb;
        float g = gate[(long)row * E_ + e];
        out[(long)row * N + col] += g * (acc[i][t][r] + bias[col]);
      }
}

// ---------------------------------------------------------------------------
// Noisy top-2 router: one wave per token
// ---------------------------------------------------------------------------
__global__ void k_router(const float* __restrict__ xn2,
                         const float* __restrict__ Wr, const float* __restrict__ br,
                         const float* __restrict__ Wn, const float* __restrict__ bn,
                         const float* __restrict__ noise, float* gate) {
  int tok = (int)((blockIdx.x * blockDim.x + threadIdx.x) >> 5);
  int l = lane_id();
  if (tok >= T_) return;
  const float* xr = xn2 + (long)tok * C_;
  float lg[E_], nl[E_];
#pragma unroll
  for (int e = 0; e < E_; ++e) {
    float s0 = 0.f, s1 = 0.f;
    for (int c = l; c < C_; c += 32) {
      float xv = xr[c];
      s0 += xv * Wr[(long)c * E_ + e];
      s1 += xv * Wn[(long)c * E_ + e];
    }
#pragma unroll
    for (int m = 16; m; m >>= 1) {
      s0 += __shfl_xor(s0, m);
      s1 += __shfl_xor(s1, m);
    }
    lg[e] = s0 + br[e];
    nl[e] = s1 + bn[e];
  }
  if (l == 0) {
    float noisy[E_];
#pragma unroll
    for (int e = 0; e < E_; ++e) {
      float z = nl[e];
      float sp = (z > 20.f) ? z : log1pf(expf(z));
      noisy[e] = lg[e] + noise[(long)tok * E_ + e] * sp;
    }
    int i0 = 0;
    for (int e = 1; e < E_; ++e) if (noisy[e] > noisy[i0]) i0 = e;
    int i1 = -1;
    for (int e = 0; e < E_; ++e) {
      if (e == i0) continue;
      if (i1 < 0 || noisy[e] > noisy[i1]) i1 = e;
    }
    float eb = expf(noisy[i1] - noisy[i0]);
    float inv = 1.f / (1.f + eb);
#pragma unroll
    for (int e = 0; e < E_; ++e) gate[(long)tok * E_ + e] = 0.f;
    gate[(long)tok * E_ + i0] = inv;
    gate[(long)tok * E_ + i1] = eb * inv;
  }
}

// ---------------------------------------------------------------------------
// Host-side orchestration
// ---------------------------------------------------------------------------
extern "C" void kernel_launch(void* const* d_in, const int* in_sizes, int n_in,
                              void* d_out, int out_size, void* d_ws, size_t ws_size,
                              hipStream_t stream) {
  const float* x     = (const float*)d_in[0];
  const float* noise = (const float*)d_in[1];
  const float* ln1_g = (const float*)d_in[2];
  const float* ln1_b = (const float*)d_in[3];
  const float* ln2_g = (const float*)d_in[4];
  const float* ln2_b = (const float*)d_in[5];
  const float* Wq    = (const float*)d_in[6];
  const float* Wk    = (const float*)d_in[7];
  const float* Wv    = (const float*)d_in[8];
  const float* Wproj = (const float*)d_in[9];
  const float* bproj = (const float*)d_in[10];
  const float* Wr    = (const float*)d_in[11];
  const float* br    = (const float*)d_in[12];
  const float* Wn    = (const float*)d_in[13];
  const float* bn    = (const float*)d_in[14];
  const float* W1    = (const float*)d_in[15];
  const float* b1    = (const float*)d_in[16];
  const float* W2    = (const float*)d_in[17];
  const float* b2    = (const float*)d_in[18];
  float* outf = (float*)d_out;

  char* wptr = (char*)d_ws;
  auto alloc = [&](size_t bytes) -> void* {
    void* p = (void*)wptr;
    wptr += (bytes + 255) & ~(size_t)255;
    return p;
  };
  __bf16* xnb    = (__bf16*)alloc((size_t)T_ * C_ * 2);
  __bf16* qb     = (__bf16*)alloc((size_t)H_ * T_ * D_ * 2);
  __bf16* kb     = (__bf16*)alloc((size_t)H_ * T_ * D_ * 2);
  __bf16* vTb    = (__bf16*)alloc((size_t)H_ * D_ * T_ * 2);
  __bf16* attb   = (__bf16*)alloc((size_t)T_ * C_ * 2);
  float*  x2f    = (float*) alloc((size_t)T_ * C_ * 4);
  float*  xn2f   = (float*) alloc((size_t)T_ * C_ * 4);
  __bf16* xn2b   = (__bf16*)alloc((size_t)T_ * C_ * 2);
  float*  gate   = (float*) alloc((size_t)T_ * E_ * 4);
  __bf16* h1b    = (__bf16*)alloc((size_t)T_ * F_ * 2);
  __bf16* WqkvT  = (__bf16*)alloc((size_t)3 * C_ * C_ * 2);
  __bf16* WprojT = (__bf16*)alloc((size_t)C_ * C_ * 2);
  __bf16* W1T    = (__bf16*)alloc((size_t)C_ * F_ * 2);
  __bf16* W2T    = (__bf16*)alloc((size_t)F_ * C_ * 2);

  // weight conversion
  k_wqkv<<<(3 * C_ * C_) / 256, 256, 0, stream>>>(Wq, Wk, Wv, WqkvT);
  k_tcvt<<<(C_ * C_) / 256, 256, 0, stream>>>(Wproj, WprojT, C_, C_);

  // LN1 -> bf16
  k_ln<<<T_, 256, 0, stream>>>(x, ln1_g, ln1_b, xnb, (float*)nullptr, C_);

  // QKV (M=T, N=3C, K=C): (T/32)*(3C/64) waves, 4 waves/block
  {
    int waves = (T_ >> 5) * ((3 * C_) >> 6);
    k_gemm_qkv<<<waves / 4, 128, 0, stream>>>(xnb, WqkvT, qb, kb, vTb,
                                              T_, 3 * C_, C_);
  }
  // attention: H * T/16 waves
  {
    int waves = H_ * (T_ >> 4);
    k_attn<<<waves / 4, 128, 0, stream>>>(qb, kb, vTb, attb);
  }
  // proj + residual (seeds d_out with x2)
  {
    int waves = (T_ >> 5) * (C_ >> 6);
    k_gemm_proj<<<waves / 4, 128, 0, stream>>>(attb, WprojT, bproj, x, x2f,
                                               outf, T_, C_, C_);
  }
  // LN2 -> bf16 + f32
  k_ln<<<T_, 256, 0, stream>>>(x2f, ln2_g, ln2_b, xn2b, xn2f, C_);

  // router (noisy top-2 gates)
  k_router<<<(T_ / 4), 128, 0, stream>>>(xn2f, Wr, br, Wn, bn, noise, gate);

  // MoE experts (dense, matches reference math; gate==0 for unselected)
  for (int e = 0; e < E_; ++e) {
    k_tcvt<<<(C_ * F_) / 256, 256, 0, stream>>>(W1 + (size_t)e * C_ * F_, W1T,
                                                C_, F_);
    int waves1 = (T_ >> 5) * (F_ >> 6);
    k_gemm_relu<<<waves1 / 4, 128, 0, stream>>>(xn2b, W1T, b1 + (size_t)e * F_,
                                                h1b, T_, F_, C_);
    k_tcvt<<<(F_ * C_) / 256, 256, 0, stream>>>(W2 + (size_t)e * F_ * C_, W2T,
                                                F_, C_);
    int waves2 = (T_ >> 5) * (C_ >> 6);
    k_gemm_moe2<<<waves2 / 4, 128, 0, stream>>>(h1b, W2T, b2 + (size_t)e * C_,
                                                gate, e, outf, T_, C_, F_);
  }
}
